// GraphAttentionLayer_36610301231189
// MI455X (gfx1250) — matrix operand order repaired and checked by reading, not compiled, not run on hardware
//
#include <hip/hip_runtime.h>
#include <hip/hip_bf16.h>

typedef __attribute__((ext_vector_type(16))) _Float16 v16h;
typedef __attribute__((ext_vector_type(8)))  float    v8f;

#define WMMA_F16(a, b, c) \
  __builtin_amdgcn_wmma_f32_16x16x32_f16(false, (a), false, (b), (short)0, (c), false, false)

constexpr int   N_NODES = 8192;
constexpr int   F_IN    = 512;
constexpr int   F_OUT   = 128;
constexpr float SLOPE   = 0.2f;
constexpr float LOG2E   = 1.4426950408889634f;

// ---------------------------------------------------------------------------
// 0) WT[n][k] = (f16) W[k][n]   (makes B fragments of the Wh GEMM contiguous)
// ---------------------------------------------------------------------------
__global__ void gat_wt_kernel(const float* __restrict__ W, _Float16* __restrict__ WT) {
  int idx = blockIdx.x * blockDim.x + threadIdx.x;   // 512*128 = 65536 threads
  int k = idx >> 7;
  int n = idx & 127;
  WT[n * F_IN + k] = (_Float16)W[idx];
}

// ---------------------------------------------------------------------------
// 1) Wh = h @ W  (fp32 out, WMMA f16).  One wave per 16x128 tile.
// ---------------------------------------------------------------------------
__global__ void __launch_bounds__(128)
gat_wh_kernel(const float* __restrict__ h, const _Float16* __restrict__ WT,
              float* __restrict__ Wh) {
  const int lane = threadIdx.x & 31;
  const int wv   = threadIdx.x >> 5;
  const int m    = lane & 15;
  const int lh   = lane >> 4;
  const int i0   = (blockIdx.x * 4 + wv) * 16;
  const int row  = i0 + m;

  v8f acc[8];
#pragma unroll
  for (int nt = 0; nt < 8; ++nt) acc[nt] = (v8f)0.0f;

  for (int kc = 0; kc < F_IN; kc += 32) {
    // A fragment: runA = kc+16*lh+[0..7], runB = kc+16*lh+8+[0..7]
    const float4* pa = (const float4*)(h + (size_t)row * F_IN + kc + 16 * lh);
    const float4* pb = (const float4*)(h + (size_t)row * F_IN + kc + 16 * lh + 8);
    float4 ra0 = pa[0], ra1 = pa[1];
    float4 rb0 = pb[0], rb1 = pb[1];
    v16h a;
    a[0]  = (_Float16)ra0.x; a[1]  = (_Float16)ra0.y; a[2]  = (_Float16)ra0.z; a[3]  = (_Float16)ra0.w;
    a[4]  = (_Float16)ra1.x; a[5]  = (_Float16)ra1.y; a[6]  = (_Float16)ra1.z; a[7]  = (_Float16)ra1.w;
    a[8]  = (_Float16)rb0.x; a[9]  = (_Float16)rb0.y; a[10] = (_Float16)rb0.z; a[11] = (_Float16)rb0.w;
    a[12] = (_Float16)rb1.x; a[13] = (_Float16)rb1.y; a[14] = (_Float16)rb1.z; a[15] = (_Float16)rb1.w;
#pragma unroll
    for (int nt = 0; nt < 8; ++nt) {
      // B fragment: lane holds N = nt*16+m, K = kc + 16*lh + [0..15] (contiguous)
      v16h b = *(const v16h*)(WT + (size_t)(nt * 16 + m) * F_IN + kc + 16 * lh);
      acc[nt] = WMMA_F16(a, b, acc[nt]);
    }
  }
#pragma unroll
  for (int nt = 0; nt < 8; ++nt)
#pragma unroll
    for (int r = 0; r < 8; ++r)
      Wh[(size_t)(i0 + r + 8 * lh) * F_OUT + nt * 16 + m] = acc[nt][r];
}

// ---------------------------------------------------------------------------
// 2) WhT[n][i] = (f16) Wh[i][n]   (B operand of attention GEMM; L2-resident)
// ---------------------------------------------------------------------------
__global__ void gat_wht_kernel(const float* __restrict__ Wh, _Float16* __restrict__ WhT) {
  __shared__ float t[16][17];
  int n = blockIdx.x * 16 + threadIdx.x;
  int i = blockIdx.y * 16 + threadIdx.y;
  t[threadIdx.y][threadIdx.x] = Wh[(size_t)i * F_OUT + n];
  __syncthreads();
  int n2 = blockIdx.x * 16 + threadIdx.y;
  int i2 = blockIdx.y * 16 + threadIdx.x;
  WhT[(size_t)n2 * N_NODES + i2] = (_Float16)t[threadIdx.x][threadIdx.y];
}

// ---------------------------------------------------------------------------
// 3) f1 = log2(e) * Wh@a[:128], f2 = log2(e) * Wh@a[128:]
//    Prescaled into the exp2 domain: lrelu(c*x) = c*lrelu(x) for c>0, so the
//    whole score pipeline (incl. the max bound) runs prescaled.
// ---------------------------------------------------------------------------
__global__ void gat_fvec_kernel(const float* __restrict__ Wh, const float* __restrict__ a,
                                float* __restrict__ f1, float* __restrict__ f2) {
  int lane = threadIdx.x & 31;
  int wv   = threadIdx.x >> 5;
  int row  = blockIdx.x * 8 + wv;
  float s1 = 0.f, s2 = 0.f;
#pragma unroll
  for (int k = 0; k < 4; ++k) {
    float v = Wh[(size_t)row * F_OUT + lane + 32 * k];
    s1 += v * a[lane + 32 * k];
    s2 += v * a[F_OUT + lane + 32 * k];
  }
#pragma unroll
  for (int off = 16; off > 0; off >>= 1) {
    s1 += __shfl_xor(s1, off, 32);
    s2 += __shfl_xor(s2, off, 32);
  }
  if (lane == 0) { f1[row] = s1 * LOG2E; f2[row] = s2 * LOG2E; }
}

// ---------------------------------------------------------------------------
// 4) f2max = max(f2)   (single block)
// ---------------------------------------------------------------------------
__global__ void gat_fmax_kernel(const float* __restrict__ f2, float* __restrict__ f2max) {
  __shared__ float red[256];
  float mx = -1e30f;
  for (int i = threadIdx.x; i < N_NODES; i += 256) mx = fmaxf(mx, f2[i]);
  red[threadIdx.x] = mx;
  __syncthreads();
  for (int s = 128; s > 0; s >>= 1) {
    if (threadIdx.x < s) red[threadIdx.x] = fmaxf(red[threadIdx.x], red[threadIdx.x + s]);
    __syncthreads();
  }
  if (threadIdx.x == 0) *f2max = red[0];
}

// ---------------------------------------------------------------------------
// 5) Fused masked-softmax + P@Wh.
//    Block = 32 rows, 8 waves split j into eighths (2048 waves total).
//    2-stage software pipeline on the adj HBM stream; B frags loaded at the
//    top of each chunk so the P-computation VALU block covers their latency.
//    Row sums via an extra WMMA against an all-ones B matrix.
// ---------------------------------------------------------------------------
struct AdjStage { int4 a0, a1, a2, a3, b0, b1, b2, b3; };

#define LOADADJ(S, JC)                                                        \
  {                                                                           \
    const int oA_ = (JC) + 16 * lh;                                           \
    const int oB_ = (JC) + 16 * lh + 8;                                       \
    S.a0 = *(const int4*)(adj + rowA + oA_);                                  \
    S.a1 = *(const int4*)(adj + rowA + oA_ + 4);                              \
    S.a2 = *(const int4*)(adj + rowA + oB_);                                  \
    S.a3 = *(const int4*)(adj + rowA + oB_ + 4);                              \
    S.b0 = *(const int4*)(adj + rowB + oA_);                                  \
    S.b1 = *(const int4*)(adj + rowB + oA_ + 4);                              \
    S.b2 = *(const int4*)(adj + rowB + oB_);                                  \
    S.b3 = *(const int4*)(adj + rowB + oB_ + 4);                              \
  }

#define ATTN_CHUNK(JC, S)                                                     \
  {                                                                           \
    const int jc_  = (JC);                                                    \
    const int offA_ = jc_ + 16 * lh;                                          \
    const int offB_ = jc_ + 16 * lh + 8;                                      \
    v16h bf[8];                                                               \
    _Pragma("unroll")                                                         \
    for (int nt = 0; nt < 8; ++nt)                                            \
      bf[nt] = *(const v16h*)(WhT + (size_t)(nt * 16 + m) * N_NODES + jc_ + 16 * lh); \
    float4 g0_ = *(const float4*)(f2 + offA_);                                \
    float4 g1_ = *(const float4*)(f2 + offA_ + 4);                            \
    float4 g2_ = *(const float4*)(f2 + offB_);                                \
    float4 g3_ = *(const float4*)(f2 + offB_ + 4);                            \
    const float fv_[16] = {g0_.x, g0_.y, g0_.z, g0_.w, g1_.x, g1_.y, g1_.z, g1_.w, \
                           g2_.x, g2_.y, g2_.z, g2_.w, g3_.x, g3_.y, g3_.z, g3_.w}; \
    const int adA_[16] = {S.a0.x, S.a0.y, S.a0.z, S.a0.w, S.a1.x, S.a1.y, S.a1.z, S.a1.w, \
                          S.a2.x, S.a2.y, S.a2.z, S.a2.w, S.a3.x, S.a3.y, S.a3.z, S.a3.w}; \
    const int adB_[16] = {S.b0.x, S.b0.y, S.b0.z, S.b0.w, S.b1.x, S.b1.y, S.b1.z, S.b1.w, \
                          S.b2.x, S.b2.y, S.b2.z, S.b2.w, S.b3.x, S.b3.y, S.b3.z, S.b3.w}; \
    v16h a0_, a1_;                                                            \
    _Pragma("unroll")                                                         \
    for (int t = 0; t < 16; ++t) {                                            \
      float e0 = f1a + fv_[t]; e0 = fmaxf(e0, SLOPE * e0);                    \
      float p0 = (adA_[t] > 0) ? exp2f(e0 - mA) : 0.0f;                       \
      a0_[t] = (_Float16)p0;                                                  \
      float e1 = f1b + fv_[t]; e1 = fmaxf(e1, SLOPE * e1);                    \
      float p1 = (adB_[t] > 0) ? exp2f(e1 - mB) : 0.0f;                       \
      a1_[t] = (_Float16)p1;                                                  \
    }                                                                         \
    rsacc0 = WMMA_F16(a0_, ones, rsacc0);                                     \
    rsacc1 = WMMA_F16(a1_, ones, rsacc1);                                     \
    _Pragma("unroll")                                                         \
    for (int nt = 0; nt < 8; ++nt) {                                          \
      acc0[nt] = WMMA_F16(a0_, bf[nt], acc0[nt]);                             \
      acc1[nt] = WMMA_F16(a1_, bf[nt], acc1[nt]);                             \
    }                                                                         \
  }

__global__ void __launch_bounds__(256)
gat_attn_kernel(const int* __restrict__ adj, const _Float16* __restrict__ WhT,
                const float* __restrict__ f1, const float* __restrict__ f2,
                const float* __restrict__ f2max_p, float* __restrict__ out) {
  __shared__ float sacc[32 * F_OUT];
  __shared__ float srs[32];
  const int tid  = threadIdx.x;
  const int lane = tid & 31;
  const int wv   = tid >> 5;
  const int m    = lane & 15;
  const int lh   = lane >> 4;
  const int i0   = blockIdx.x * 32;

  for (int idx = tid; idx < 32 * F_OUT; idx += 256) sacc[idx] = 0.0f;
  if (tid < 32) srs[tid] = 0.0f;
  __syncthreads();

  const float f2m = *f2max_p;                  // prescaled domain
  const float f1a = f1[i0 + m];
  const float f1b = f1[i0 + 16 + m];
  float mA = f1a + f2m; mA = fmaxf(mA, SLOPE * mA);  // lrelu upper bound (prescaled)
  float mB = f1b + f2m; mB = fmaxf(mB, SLOPE * mB);

  const size_t rowA = (size_t)(i0 + m) * N_NODES;
  const size_t rowB = (size_t)(i0 + 16 + m) * N_NODES;

  v16h ones;
#pragma unroll
  for (int t = 0; t < 16; ++t) ones[t] = (_Float16)1.0f;

  v8f acc0[8], acc1[8];
#pragma unroll
  for (int nt = 0; nt < 8; ++nt) { acc0[nt] = (v8f)0.0f; acc1[nt] = (v8f)0.0f; }
  v8f rsacc0 = (v8f)0.0f, rsacc1 = (v8f)0.0f;

  const int jbeg = wv * (N_NODES / 8);
  const int jend = jbeg + (N_NODES / 8);     // 1024 j's -> 16 outer iterations

  AdjStage sA, sB;
  LOADADJ(sA, jbeg);
  for (int jc = jbeg; jc < jend; jc += 64) {
    LOADADJ(sB, jc + 32);                    // stage next half-chunk (HBM)
    if (jc + 544 <= jend) {                  // stream adj further ahead
      __builtin_prefetch(adj + rowA + jc + 512, 0, 0);
      __builtin_prefetch(adj + rowB + jc + 512, 0, 0);
    }
    ATTN_CHUNK(jc, sA);                      // compute covers sB load latency
    const int jn = (jc + 64 < jend) ? (jc + 64) : jbeg;  // dummy reload at tail
    LOADADJ(sA, jn);
    ATTN_CHUNK(jc + 32, sB);                 // compute covers sA load latency
  }

  // combine j-split partials (additive thanks to fixed per-row max)
#pragma unroll
  for (int nt = 0; nt < 8; ++nt)
#pragma unroll
    for (int r = 0; r < 8; ++r) {
      atomicAdd(&sacc[(r + 8 * lh) * F_OUT + nt * 16 + m], acc0[nt][r]);
      atomicAdd(&sacc[(16 + r + 8 * lh) * F_OUT + nt * 16 + m], acc1[nt][r]);
    }
  if (m == 0) {  // row sums replicated across all 16 columns; take column 0
#pragma unroll
    for (int r = 0; r < 8; ++r) {
      atomicAdd(&srs[r + 8 * lh], rsacc0[r]);
      atomicAdd(&srs[16 + r + 8 * lh], rsacc1[r]);
    }
  }
  __syncthreads();

  for (int idx = tid; idx < 32 * F_OUT; idx += 256) {
    int r = idx >> 7;
    float s = srs[r];
    float inv = s > 0.f ? 1.0f / s : 0.0f;
    out[(size_t)(i0 + r) * F_OUT + (idx & 127)] = sacc[idx] * inv;
  }
}

// ---------------------------------------------------------------------------
extern "C" void kernel_launch(void* const* d_in, const int* in_sizes, int n_in,
                              void* d_out, int out_size, void* d_ws, size_t ws_size,
                              hipStream_t stream) {
  (void)in_sizes; (void)n_in; (void)out_size; (void)ws_size;
  const float* h   = (const float*)d_in[0];
  const int*   adj = (const int*)d_in[1];
  const float* W   = (const float*)d_in[2];
  const float* a   = (const float*)d_in[3];
  float* out = (float*)d_out;

  char* ws = (char*)d_ws;
  float*    Wh   = (float*)(ws);                                 // 4 MB
  _Float16* WhT  = (_Float16*)(ws + (4u << 20));                 // 2 MB
  _Float16* WT   = (_Float16*)(ws + (6u << 20));                 // 128 KB
  float*    f1   = (float*)(ws + (6u << 20) + (128u << 10));     // 32 KB
  float*    f2   = f1 + N_NODES;                                 // 32 KB
  float*    f2mx = f2 + N_NODES;                                 // 4 B

  gat_wt_kernel<<<256, 256, 0, stream>>>(W, WT);
  gat_wh_kernel<<<128, 128, 0, stream>>>(h, WT, Wh);
  gat_wht_kernel<<<dim3(8, 512), dim3(16, 16), 0, stream>>>(Wh, WhT);
  gat_fvec_kernel<<<1024, 256, 0, stream>>>(Wh, a, f1, f2);
  gat_fmax_kernel<<<1, 256, 0, stream>>>(f2, f2mx);
  gat_attn_kernel<<<256, 256, 0, stream>>>(adj, WhT, f1, f2, f2mx, out);
}
